// AttentionText_39032662786399
// MI455X (gfx1250) — compile-verified, matrix operand order checked by reading, not run on hardware
//
#include <hip/hip_runtime.h>

typedef _Float16 v16h __attribute__((ext_vector_type(16)));
typedef _Float16 v8h  __attribute__((ext_vector_type(8)));
typedef _Float16 v4h  __attribute__((ext_vector_type(4)));
typedef float    v8f  __attribute__((ext_vector_type(8)));

#define BATCH 8
#define SEQ   1024
#define CH    1024
#define HEADS 16
#define HDIM  64
#define QK_SCALE 0.125f   // 64^-0.5

// ---------------------------------------------------------------------------
// WMMA helpers (CDNA5 / gfx1250, wave32)
// ---------------------------------------------------------------------------
__device__ __forceinline__ v8f wmma32(v16h a, v16h b, v8f c) {
  // D = A(16x32 f16) * B(32x16 f16) + C(16x16 f32)
  return __builtin_amdgcn_wmma_f32_16x16x32_f16(false, a, false, b, (short)0, c,
                                                false, false);
}

// Fragment load: per lane, 8 contiguous halves at p and 8 at p+16 (elements).
// 16-bit A/B operand layout: lanes 0-15 carry K {0-7,16-23}, lanes 16-31 carry
// K {8-15,24-31} when the caller offsets p by (lane>>4)*8.
__device__ __forceinline__ v16h ldfrag(const _Float16* p) {
  v8h lo = *(const v8h*)(p);
  v8h hi = *(const v8h*)(p + 16);
  return __builtin_shufflevector(lo, hi, 0, 1, 2, 3, 4, 5, 6, 7,
                                 8, 9, 10, 11, 12, 13, 14, 15);
}

__device__ __forceinline__ v8f zero8f() {
  v8f z = {0.f, 0.f, 0.f, 0.f, 0.f, 0.f, 0.f, 0.f};
  return z;
}

// ---------------------------------------------------------------------------
// fp32 -> f16 conversion (4-wide)
// ---------------------------------------------------------------------------
__global__ void cvt_f32_f16(const float4* __restrict__ src,
                            v4h* __restrict__ dst, int n4) {
  int i = blockIdx.x * blockDim.x + threadIdx.x;
  if (i < n4) {
    float4 f = src[i];
    v4h h = {(_Float16)f.x, (_Float16)f.y, (_Float16)f.z, (_Float16)f.w};
    dst[i] = h;
  }
}

// ---------------------------------------------------------------------------
// WMMA GEMM: out(M x Nout) = A(M x K) * W^T + bias, A/W in f16 row-major.
// Block = 256 threads (8 waves), tile 128(M) x 128(N), K stepped by 32.
// Wave (wm,wn) computes a 32x64 sub-tile: 2 A-frags x 4 B-frags = 8 WMMA/step.
// mode 0: QKV epilogue -> split q (scaled), k, vT as f16
// mode 1: plain fp32 epilogue -> outf
// ---------------------------------------------------------------------------
__global__ __launch_bounds__(256) void gemm_wmma(
    const _Float16* __restrict__ A, const _Float16* __restrict__ W,
    const float* __restrict__ bias, _Float16* __restrict__ qh,
    _Float16* __restrict__ kh, _Float16* __restrict__ vTh,
    float* __restrict__ outf, int M, int K, int Nout, int mode) {
  __shared__ _Float16 As[128 * 32];   // 8 KB
  __shared__ _Float16 Bs[128 * 32];   // 8 KB

  const int tid  = threadIdx.x;
  const int lane = tid & 31;
  const int wave = tid >> 5;
  const int wm = wave & 3;            // 0..3 -> 32-row slice
  const int wn = wave >> 2;           // 0..1 -> 64-col slice
  const int ntiles = Nout >> 7;
  const int mtile = (int)blockIdx.x / ntiles;
  const int ntile = (int)blockIdx.x % ntiles;
  const int m0 = mtile * 128;
  const int n0 = ntile * 128;

  const int cr = tid >> 1;            // staging row 0..127
  const int cc = (tid & 1) * 16;      // staging col 0 / 16
  const int lrow  = lane & 15;
  const int lhalf = (lane >> 4) * 8;

  v8f acc[2][4];
#pragma unroll
  for (int i = 0; i < 2; i++)
#pragma unroll
    for (int t = 0; t < 4; t++) acc[i][t] = zero8f();

  const _Float16* agp = &A[(size_t)(m0 + cr) * K + cc];
  const _Float16* wgp = &W[(size_t)(n0 + cr) * K + cc];

  for (int k0 = 0; k0 < K; k0 += 32) {
    __syncthreads();
    *(v8h*)&As[cr * 32 + cc]     = *(const v8h*)(agp + k0);
    *(v8h*)&As[cr * 32 + cc + 8] = *(const v8h*)(agp + k0 + 8);
    *(v8h*)&Bs[cr * 32 + cc]     = *(const v8h*)(wgp + k0);
    *(v8h*)&Bs[cr * 32 + cc + 8] = *(const v8h*)(wgp + k0 + 8);
    if (k0 + 32 < K) {   // gfx1250 global_prefetch for the next K-step
      __builtin_prefetch(agp + k0 + 32, 0, 1);
      __builtin_prefetch(wgp + k0 + 32, 0, 1);
    }
    __syncthreads();

    v16h a0 = ldfrag(&As[(wm * 32 + lrow) * 32 + lhalf]);
    v16h a1 = ldfrag(&As[(wm * 32 + 16 + lrow) * 32 + lhalf]);
#pragma unroll
    for (int t = 0; t < 4; t++) {
      v16h bfrag = ldfrag(&Bs[(wn * 64 + t * 16 + lrow) * 32 + lhalf]);
      acc[0][t] = wmma32(a0, bfrag, acc[0][t]);
      acc[1][t] = wmma32(a1, bfrag, acc[1][t]);
    }
  }

  // Epilogue. C/D layout: lane<16 -> rows 0..7 (vgpr r), col lane;
  // lane>=16 -> rows 8..15, col lane-16.
#pragma unroll
  for (int i = 0; i < 2; i++) {
#pragma unroll
    for (int t = 0; t < 4; t++) {
      const int col = n0 + wn * 64 + t * 16 + lrow;
      const float bv = bias[col];
#pragma unroll
      for (int r = 0; r < 8; r++) {
        const int row = m0 + wm * 32 + i * 16 + lhalf + r;
        const float val = acc[i][t][r] + bv;
        if (mode == 1) {
          outf[(size_t)row * Nout + col] = val;
        } else {
          const int b = row >> 10;     // M = BATCH*SEQ
          const int n = row & 1023;
          if (col < CH) {              // Q, pre-scaled
            const int h = col >> 6, d = col & 63;
            qh[(((size_t)(b * HEADS + h) * SEQ) + n) * HDIM + d] =
                (_Float16)(val * QK_SCALE);
          } else if (col < 2 * CH) {   // K
            const int c2 = col - CH;
            const int h = c2 >> 6, d = c2 & 63;
            kh[(((size_t)(b * HEADS + h) * SEQ) + n) * HDIM + d] = (_Float16)val;
          } else {                     // V stored transposed: (b,h,d,n)
            const int c2 = col - 2 * CH;
            const int h = c2 >> 6, d = c2 & 63;
            vTh[(((size_t)(b * HEADS + h) * HDIM) + d) * SEQ + n] = (_Float16)val;
          }
        }
      }
    }
  }
}

// ---------------------------------------------------------------------------
// Attention: per (b,h), S = Q K^T (scale folded into Q), softmax, O = P V.
// Block = 128 threads (4 waves); each wave owns 16 query rows. K/V chunks
// (32 keys) are staged in LDS once per block and shared by all 4 waves.
// S^T tiles via wmma(K_tile, Q^T): each lane holds 8 scores of one query ->
// softmax reduction is 8 in-lane values + one shfl_xor(16).
// ---------------------------------------------------------------------------
__global__ __launch_bounds__(128) void attn_wmma(
    const _Float16* __restrict__ qh, const _Float16* __restrict__ kh,
    const _Float16* __restrict__ vTh, _Float16* __restrict__ ctx) {
  __shared__ _Float16 Ks[32 * 64];     // 32 keys x 64 dims   (4 KB)
  __shared__ _Float16 Vs[64 * 32];     // 64 dims x 32 keys   (4 KB)
  __shared__ _Float16 P[4][16 * 32];   // per-wave P staging  (4 KB)

  const int tid  = threadIdx.x;
  const int lane = tid & 31;
  const int wave = tid >> 5;
  const int bh     = (int)blockIdx.x >> 4;   // 0..127
  const int rowblk = (int)blockIdx.x & 15;
  const int q0 = rowblk * 64 + wave * 16;
  const int b = bh >> 4, h = bh & 15;

  const _Float16* Q  = qh  + (size_t)bh * SEQ * HDIM;
  const _Float16* Km = kh  + (size_t)bh * SEQ * HDIM;
  const _Float16* VT = vTh + (size_t)bh * HDIM * SEQ;

  const int lrow  = lane & 15;
  const int lhalf = (lane >> 4) * 8;

  // staging maps
  const int kr = tid >> 2, kc = (tid & 3) * 16;   // Ks: 32 rows x 64 halves
  const int vr = tid >> 1, vc = (tid & 1) * 16;   // Vs: 64 rows x 32 halves

  // Q as B-operand (Q rows load directly in B-fragment layout)
  const _Float16* qp = Q + (size_t)(q0 + lrow) * HDIM + lhalf;
  const v16h qb0 = ldfrag(qp);        // dims 0..31
  const v16h qb1 = ldfrag(qp + 32);   // dims 32..63

  // ---- Pass 1: online row max / sum (per-lane half, merged at the end) ----
  float m_run = -3.0e38f, s_run = 0.f;
  for (int j2 = 0; j2 < SEQ / 32; j2++) {
    __syncthreads();
    *(v8h*)&Ks[kr * 64 + kc]     = *(const v8h*)&Km[(size_t)(j2 * 32 + kr) * HDIM + kc];
    *(v8h*)&Ks[kr * 64 + kc + 8] = *(const v8h*)&Km[(size_t)(j2 * 32 + kr) * HDIM + kc + 8];
    __syncthreads();
#pragma unroll
    for (int t = 0; t < 2; t++) {
      v16h ka0 = ldfrag(&Ks[(t * 16 + lrow) * 64 + lhalf]);
      v16h ka1 = ldfrag(&Ks[(t * 16 + lrow) * 64 + 32 + lhalf]);
      v8f s = zero8f();
      s = wmma32(ka0, qb0, s);
      s = wmma32(ka1, qb1, s);
      float mt = m_run;
#pragma unroll
      for (int r = 0; r < 8; r++) mt = fmaxf(mt, s[r]);
      const float rescale = __expf(m_run - mt);
      float ps = 0.f;
#pragma unroll
      for (int r = 0; r < 8; r++) ps += __expf(s[r] - mt);
      s_run = s_run * rescale + ps;
      m_run = mt;
    }
  }
  const float m_o = __shfl_xor(m_run, 16, 32);
  const float s_o = __shfl_xor(s_run, 16, 32);
  const float m_tot = fmaxf(m_run, m_o);
  const float s_tot = s_run * __expf(m_run - m_tot) + s_o * __expf(m_o - m_tot);
  const float inv_s = 1.0f / s_tot;

  // ---- Pass 2: P = softmax(S), O += P * V (32 keys per step) ----
  v8f o[4];
#pragma unroll
  for (int t = 0; t < 4; t++) o[t] = zero8f();

  for (int j2 = 0; j2 < SEQ / 32; j2++) {
    __syncthreads();
    *(v8h*)&Ks[kr * 64 + kc]     = *(const v8h*)&Km[(size_t)(j2 * 32 + kr) * HDIM + kc];
    *(v8h*)&Ks[kr * 64 + kc + 8] = *(const v8h*)&Km[(size_t)(j2 * 32 + kr) * HDIM + kc + 8];
    *(v8h*)&Vs[vr * 32 + vc]     = *(const v8h*)&VT[(size_t)vr * SEQ + j2 * 32 + vc];
    *(v8h*)&Vs[vr * 32 + vc + 8] = *(const v8h*)&VT[(size_t)vr * SEQ + j2 * 32 + vc + 8];
    __syncthreads();
#pragma unroll
    for (int t = 0; t < 2; t++) {
      v16h ka0 = ldfrag(&Ks[(t * 16 + lrow) * 64 + lhalf]);
      v16h ka1 = ldfrag(&Ks[(t * 16 + lrow) * 64 + 32 + lhalf]);
      v8f s = zero8f();
      s = wmma32(ka0, qb0, s);
      s = wmma32(ka1, qb1, s);
#pragma unroll
      for (int r = 0; r < 8; r++) {
        const float pv = __expf(s[r] - m_tot) * inv_s;
        P[wave][lrow * 32 + t * 16 + lhalf + r] = (_Float16)pv;
      }
    }
    // P tile back as A-fragment (same-wave LDS RAW: DS ops are in-order)
    v16h pa = ldfrag(&P[wave][lrow * 32 + lhalf]);
#pragma unroll
    for (int t = 0; t < 4; t++) {
      v16h vb = ldfrag(&Vs[(t * 16 + lrow) * 32 + lhalf]);
      o[t] = wmma32(pa, vb, o[t]);
    }
  }

  // ---- Epilogue: ctx (B, N, C) f16 ----
#pragma unroll
  for (int t = 0; t < 4; t++) {
    const int d = t * 16 + lrow;
#pragma unroll
    for (int r = 0; r < 8; r++) {
      const int qn = q0 + lhalf + r;
      ctx[((size_t)b * SEQ + qn) * CH + h * HDIM + d] = (_Float16)o[t][r];
    }
  }
}

// ---------------------------------------------------------------------------
// Host launch
// ---------------------------------------------------------------------------
extern "C" void kernel_launch(void* const* d_in, const int* in_sizes, int n_in,
                              void* d_out, int out_size, void* d_ws,
                              size_t ws_size, hipStream_t stream) {
  const float* x  = (const float*)d_in[0];
  const float* w1 = (const float*)d_in[1];   // (3C, C)
  const float* b1 = (const float*)d_in[2];   // (3C)
  const float* w2 = (const float*)d_in[3];   // (C, C)
  const float* b2 = (const float*)d_in[4];   // (C)
  float* out = (float*)d_out;

  const size_t M   = (size_t)BATCH * SEQ;      // 8192
  const size_t nX  = M * CH;                   // 8M
  const size_t nW1 = (size_t)3 * CH * CH;      // 3M
  const size_t nW2 = (size_t)CH * CH;          // 1M
  const size_t nHd = (size_t)BATCH * HEADS * SEQ * HDIM;  // 8M

  _Float16* ws = (_Float16*)d_ws;
  size_t off = 0;
  _Float16* xh  = ws + off; off += nX;
  _Float16* w1h = ws + off; off += nW1;
  _Float16* w2h = ws + off; off += nW2;
  _Float16* qh  = ws + off; off += nHd;
  _Float16* kh  = ws + off; off += nHd;
  _Float16* vTh = ws + off; off += nHd;
  _Float16* ctx = ws + off; off += nX;

  // fp32 -> f16 staging (4-wide)
  cvt_f32_f16<<<(int)((nX / 4 + 255) / 256), 256, 0, stream>>>(
      (const float4*)x, (v4h*)xh, (int)(nX / 4));
  cvt_f32_f16<<<(int)((nW1 / 4 + 255) / 256), 256, 0, stream>>>(
      (const float4*)w1, (v4h*)w1h, (int)(nW1 / 4));
  cvt_f32_f16<<<(int)((nW2 / 4 + 255) / 256), 256, 0, stream>>>(
      (const float4*)w2, (v4h*)w2h, (int)(nW2 / 4));

  // QKV projection: (8192 x 1024) * (3072 x 1024)^T  -> q (scaled), k, vT
  gemm_wmma<<<(int)((M / 128) * (3 * CH / 128)), 256, 0, stream>>>(
      xh, w1h, b1, qh, kh, vTh, nullptr, (int)M, CH, 3 * CH, 0);

  // Attention: 128 (b,h) pairs x 16 row-blocks of 64 queries
  attn_wmma<<<BATCH * HEADS * (SEQ / 64), 128, 0, stream>>>(qh, kh, vTh, ctx);

  // Output projection: (8192 x 1024) * (1024 x 1024)^T + bias -> fp32 out
  gemm_wmma<<<(int)((M / 128) * (CH / 128)), 256, 0, stream>>>(
      ctx, w2h, b2, nullptr, nullptr, nullptr, out, (int)M, CH, CH, 1);
}
